// MemoryBlock_12936441496249
// MI455X (gfx1250) — matrix operand order repaired and checked by reading, not compiled, not run on hardware
//
#include <hip/hip_runtime.h>
#include <hip/hip_bf16.h>

// Problem constants (from reference)
#define B_ROWS   32768
#define MEM_DIM  2048
#define Z_DIM    256
#define TEM_INV  2.0f          // 1/0.5
#define LAMBDA   0.005f
#define EPS_SHRK 1e-12f
#define EPS_NORM 1e-12f

#define ROWS_PER_BLK 16
#define THREADS      256       // 8 waves of 32
#define ATT_STRIDE   2052      // 2048 + 4 pad (f32) -> conflict-free ds_b128 A reads

typedef __bf16 v16bf __attribute__((ext_vector_type(16)));
typedef float  v8f   __attribute__((ext_vector_type(8)));

union Frag16 {              // 16 bf16 = 32B per lane = 8 VGPRs
    v16bf v;
    uint4 q[2];
};

// ---------------------------------------------------------------------------
// Prep: convert mem (f32, [2048][256]) to bf16 row-major and bf16 transposed.
// 2 MB total, fully L2-resident afterwards. Negligible cost.
// ---------------------------------------------------------------------------
__global__ void memblk_prep_kernel(const float* __restrict__ mem,
                                   __bf16* __restrict__ memb,   // [2048][256]
                                   __bf16* __restrict__ memT) { // [256][2048]
    int m = blockIdx.x;
    int z = threadIdx.x;
    float v = mem[m * Z_DIM + z];
    __bf16 b = (__bf16)v;
    memb[m * Z_DIM + z] = b;
    memT[z * MEM_DIM + m] = b;
}

// ---------------------------------------------------------------------------
// Fused kernel: 16 rows per block, 8 waves.
//   GEMM1 (bf16 WMMA): logits 16x2048 -> LDS (f32), 2 tiles in flight/wave
//   softmax * 2 (temperature), hard-shrink-relu, L1 norm scale -> LDS
//   GEMM2 (bf16 WMMA): out 16x256 -> global
// ---------------------------------------------------------------------------
__global__ __launch_bounds__(THREADS)
void memblk_fused_kernel(const float*  __restrict__ x,     // [32768][256]
                         const __bf16* __restrict__ memb,  // [2048][256]
                         const __bf16* __restrict__ memT,  // [256][2048]
                         float*        __restrict__ out) { // [32768][256]
    __shared__ __bf16 xlds[ROWS_PER_BLK * Z_DIM];        // 8 KB
    __shared__ float  att[ROWS_PER_BLK * ATT_STRIDE];    // ~128 KB
    __shared__ float  rowscale[ROWS_PER_BLK];            // 1/L1 per row

    const int tid  = threadIdx.x;
    const int wave = tid >> 5;
    const int lane = tid & 31;
    const int l15  = lane & 15;
    const bool lo  = (lane < 16);
    const int rowbase = blockIdx.x * ROWS_PER_BLK;

    // ---- Stage x tile -> bf16 LDS (coalesced) ----
    for (int e = tid; e < ROWS_PER_BLK * Z_DIM; e += THREADS) {
        int r = e >> 8;          // /256
        int c = e & 255;
        xlds[e] = (__bf16)x[(size_t)(rowbase + r) * Z_DIM + c];
    }
    __syncthreads();

    // =====================================================================
    // GEMM1: att[16][2048] = xlds[16][256] * memb^T   (wave owns 256 cols)
    // Two N-tiles interleaved -> two independent WMMA accumulator chains.
    // =====================================================================
    {
        // Preload all 8 A fragments (K=256 -> 8 chunks of 32)
        // A layout (16-bit 16x32): lanes 0-15 -> M=lane, K {0..7,16..23};
        // lanes 16-31 -> M=lane-16, K {8..15,24..31}
        Frag16 af[8];
        const int akoff = lo ? 0 : 8;
#pragma unroll
        for (int kc = 0; kc < 8; ++kc) {
            const __bf16* pa = &xlds[l15 * Z_DIM + kc * 32 + akoff];
            af[kc].q[0] = *(const uint4*)(pa);
            af[kc].q[1] = *(const uint4*)(pa + 16);
        }

        const int nbase = wave * 256;
        // B layout (16-bit 32x16): lanes 0-15 -> N=lane, K {0..7,16..23};
        // lanes 16-31 -> N=lane-16, K {8..15,24..31}
        const int bkoff = lo ? 0 : 8;
        const int cr0   = lo ? 0 : 8;

        for (int nt = 0; nt < 16; nt += 2) {
            const int mrow0 = nbase + nt * 16 + l15;        // mem rows (= cols)
            const __bf16* pb0 = memb + (size_t)mrow0 * Z_DIM + bkoff;
            const __bf16* pb1 = pb0 + 16 * Z_DIM;

            v8f acc0, acc1;
#pragma unroll
            for (int i = 0; i < 8; ++i) { acc0[i] = 0.0f; acc1[i] = 0.0f; }

#pragma unroll
            for (int kc = 0; kc < 8; ++kc) {
                Frag16 b0, b1;
                b0.q[0] = *(const uint4*)(pb0 + kc * 32);
                b0.q[1] = *(const uint4*)(pb0 + kc * 32 + 16);
                b1.q[0] = *(const uint4*)(pb1 + kc * 32);
                b1.q[1] = *(const uint4*)(pb1 + kc * 32 + 16);
                acc0 = __builtin_amdgcn_wmma_f32_16x16x32_bf16(
                        false, af[kc].v, false, b0.v,
                        (short)0, acc0, false, false);
                acc1 = __builtin_amdgcn_wmma_f32_16x16x32_bf16(
                        false, af[kc].v, false, b1.v,
                        (short)0, acc1, false, false);
            }

            const int cn = nbase + nt * 16 + l15;
#pragma unroll
            for (int r = 0; r < 8; ++r) {
                att[(cr0 + r) * ATT_STRIDE + cn]      = acc0[r];
                att[(cr0 + r) * ATT_STRIDE + cn + 16] = acc1[r];
            }
        }
    }
    __syncthreads();

    // =====================================================================
    // softmax(att*2) -> hard_shrink_relu -> store shrunk, rowscale = 1/L1
    // wave w owns rows 2w..2w+1; lanes stride columns (bank-friendly)
    // =====================================================================
    for (int r = wave * 2; r < wave * 2 + 2; ++r) {
        float* row = &att[r * ATT_STRIDE];

        // online max / sum-exp
        float m = -3.0e38f, s = 0.0f;
        for (int c = lane; c < MEM_DIM; c += 32) {
            float l = TEM_INV * row[c];
            float nm = fmaxf(m, l);
            s = s * __expf(m - nm) + __expf(l - nm);
            m = nm;
        }
#pragma unroll
        for (int off = 16; off > 0; off >>= 1) {
            float om = __shfl_xor(m, off, 32);
            float os = __shfl_xor(s, off, 32);
            float nm = fmaxf(m, om);
            s = s * __expf(m - nm) + os * __expf(om - nm);
            m = nm;
        }
        const float rsum = 1.0f / s;

        // p -> g(p) = relu(p - lambda) * p / (|p - lambda| + eps); L1 accumulate
        float l1 = 0.0f;
        for (int c = lane; c < MEM_DIM; c += 32) {
            float p = __expf(TEM_INV * row[c] - m) * rsum;
            float d = p - LAMBDA;
            float g = (d > 0.0f ? d : 0.0f) * p / (fabsf(d) + EPS_SHRK);
            row[c] = g;
            l1 += g;    // g >= 0
        }
#pragma unroll
        for (int off = 16; off > 0; off >>= 1)
            l1 += __shfl_xor(l1, off, 32);

        if (lane == 0)
            rowscale[r] = 1.0f / fmaxf(l1, EPS_NORM);
    }
    __syncthreads();

    // =====================================================================
    // GEMM2: out[16][256] = (att * rowscale)[16][2048] * memT^T
    // wave owns 32 output cols (2 N-tiles); K = 2048 -> 64 steps of 32
    // =====================================================================
    {
        const int z0 = wave * 32;
        const float ascale = rowscale[l15];
        const int akoff = lo ? 0 : 8;
        const int bkoff = lo ? 0 : 8;

        v8f acc[2];
#pragma unroll
        for (int nt = 0; nt < 2; ++nt)
#pragma unroll
            for (int i = 0; i < 8; ++i) acc[nt][i] = 0.0f;

        for (int kc = 0; kc < MEM_DIM; kc += 32) {
            // Build A fragment from f32 LDS att, scaled by 1/L1, cvt -> bf16
            const float* pa = &att[l15 * ATT_STRIDE + kc + akoff];
            float tmp[16];
#pragma unroll
            for (int i = 0; i < 8; ++i) tmp[i] = pa[i];
#pragma unroll
            for (int i = 0; i < 8; ++i) tmp[8 + i] = pa[16 + i];
            Frag16 a2;
#pragma unroll
            for (int i = 0; i < 16; ++i)
                a2.v[i] = (__bf16)(tmp[i] * ascale);

#pragma unroll
            for (int nt = 0; nt < 2; ++nt) {
                const int zc = z0 + nt * 16 + l15;            // output col
                const __bf16* pb = memT + (size_t)zc * MEM_DIM + kc + bkoff;
                Frag16 bf;
                bf.q[0] = *(const uint4*)(pb);
                bf.q[1] = *(const uint4*)(pb + 16);
                acc[nt] = __builtin_amdgcn_wmma_f32_16x16x32_bf16(
                            false, a2.v, false, bf.v,
                            (short)0, acc[nt], false, false);
            }
        }

        const int orow0 = rowbase + (lo ? 0 : 8);
#pragma unroll
        for (int nt = 0; nt < 2; ++nt) {
            const int on = z0 + nt * 16 + l15;
#pragma unroll
            for (int r = 0; r < 8; ++r)
                out[(size_t)(orow0 + r) * Z_DIM + on] = acc[nt][r];
        }
    }
}

// ---------------------------------------------------------------------------
extern "C" void kernel_launch(void* const* d_in, const int* in_sizes, int n_in,
                              void* d_out, int out_size, void* d_ws, size_t ws_size,
                              hipStream_t stream) {
    const float* x   = (const float*)d_in[0];   // [32768*256]
    const float* mem = (const float*)d_in[1];   // [2048*256]
    float* out = (float*)d_out;                 // [32768*256]

    __bf16* memb = (__bf16*)d_ws;                       // 1 MB
    __bf16* memT = memb + (size_t)MEM_DIM * Z_DIM;      // 1 MB

    memblk_prep_kernel<<<dim3(MEM_DIM), dim3(Z_DIM), 0, stream>>>(mem, memb, memT);
    memblk_fused_kernel<<<dim3(B_ROWS / ROWS_PER_BLK), dim3(THREADS), 0, stream>>>(
        x, memb, memT, out);
}
